// simply_attention_block_63316407877678
// MI455X (gfx1250) — matrix operand order repaired and checked by reading, not compiled, not run on hardware
//
#include <hip/hip_runtime.h>
#include <hip/hip_bf16.h>

typedef __attribute__((ext_vector_type(16))) __bf16 v16bf;
typedef __attribute__((ext_vector_type(8)))  float  v8f;
typedef __attribute__((ext_vector_type(4)))  unsigned int u32x4;
typedef __attribute__((ext_vector_type(8)))  int  i32x8;
typedef __attribute__((ext_vector_type(4)))  int  i32x4;

union A16 { v16bf v; uint4 q[2]; };
union F8  { v8f  v; float f[8]; };

static __device__ __forceinline__ unsigned short f2b(float f) {
    union { float f; unsigned u; } x; x.f = f;
    unsigned r = x.u + 0x7FFFu + ((x.u >> 16) & 1u);   // round-to-nearest-even
    return (unsigned short)(r >> 16);
}

static __device__ __forceinline__ v8f wmma_bf16(v16bf a, v16bf b, v8f c) {
    return __builtin_amdgcn_wmma_f32_16x16x32_bf16(false, a, false, b, (short)0, c, false, false);
}

#define BATCH 8
#define SDIM  4096
#define DIN   512
#define HID   256

// ---------------------------------------------------------------------------
// Kernel 0: W [512][256] f32  ->  WT [256][512] bf16 (B-operand friendly)
// ---------------------------------------------------------------------------
__global__ __launch_bounds__(256) void k_wt(const float* __restrict__ W,
                                            unsigned short* __restrict__ wt) {
    int i = blockIdx.x * 256 + threadIdx.x;   // 512*256 = 131072 elements
    if (i < DIN * HID) {
        int d = i / HID, c = i % HID;
        wt[c * DIN + d] = f2b(W[i]);
    }
}

// ---------------------------------------------------------------------------
// Kernel 1: lin = x @ W + b  (bf16 WMMA, f32 accumulate)
// grid = B*S/16 workgroups of 128 threads; wave w covers columns [w*64, w*64+64)
// ---------------------------------------------------------------------------
__global__ __launch_bounds__(128) void k_linear(const float* __restrict__ x,
                                                const unsigned short* __restrict__ wt,
                                                const float* __restrict__ bias,
                                                float* __restrict__ linf,
                                                unsigned short* __restrict__ linb) {
    __shared__ __attribute__((aligned(16))) unsigned short xs[16 * DIN];  // 16 KB bf16 tile

    const int t    = threadIdx.x;
    const int lane = t & 31;
    const int w    = t >> 5;
    const int half = lane >> 4;
    const int ln16 = lane & 15;
    const long rowbase = (long)blockIdx.x * 16;

    // x tile (16 x 512 f32) -> bf16 LDS, coalesced float4 loads
    for (int i = 0; i < 16; ++i) {
        int e4 = i * 128 + t;                // 2048 float4 total
        int r  = e4 >> 7;
        int c4 = (e4 & 127) << 2;
        const float4 v = *(const float4*)&x[(rowbase + r) * DIN + c4];
        uint2 p;
        p.x = (unsigned)f2b(v.x) | ((unsigned)f2b(v.y) << 16);
        p.y = (unsigned)f2b(v.z) | ((unsigned)f2b(v.w) << 16);
        *(uint2*)&xs[r * DIN + c4] = p;
    }
    __syncthreads();

    for (int nt = 0; nt < 4; ++nt) {
        const int n0  = w * 64 + nt * 16;
        const int col = n0 + ln16;
        const float bv = bias[col];
        F8 acc;
        #pragma unroll
        for (int r = 0; r < 8; ++r) acc.f[r] = bv;

        #pragma unroll
        for (int kk = 0; kk < 16; ++kk) {
            const int kbase = kk * 32;
            A16 a, bm;
            // A: row = ln16, K groups {kbase+8h..+7, kbase+16+8h..+7}
            a.q[0] = *(const uint4*)&xs[ln16 * DIN + kbase + half * 8];
            a.q[1] = *(const uint4*)&xs[ln16 * DIN + kbase + 16 + half * 8];
            // B: col fixed per lane, K groups {kbase+8h..+7, kbase+16+8h..+7}
            bm.q[0] = *(const uint4*)&wt[col * DIN + kbase + half * 8];
            bm.q[1] = *(const uint4*)&wt[col * DIN + kbase + 16 + half * 8];
            acc.v = wmma_bf16(a.v, bm.v, acc.v);
        }

        #pragma unroll
        for (int r = 0; r < 8; ++r) {
            long grow = rowbase + r + half * 8;
            linf[grow * HID + col] = acc.f[r];
            linb[grow * HID + col] = f2b(acc.f[r]);
        }
    }
}

// ---------------------------------------------------------------------------
// Kernel 2: fused flash-attention (Q=K=V=lin) + residual + LayerNorm
// grid = B * (S/64) workgroups of 128 threads; each wave owns 16 query rows.
// K/V tiles are staged into LDS by the Tensor Data Mover (tensor_load_to_lds).
// ---------------------------------------------------------------------------
__global__ __launch_bounds__(128) void k_attn(const float* __restrict__ linf,
                                              const unsigned short* __restrict__ linb,
                                              const float* __restrict__ gamma,
                                              const float* __restrict__ beta,
                                              float* __restrict__ out) {
    __shared__ __attribute__((aligned(16))) unsigned short kv [32 * HID];      // 16 KB: K/V tile (row major, TDM dest)
    __shared__ __attribute__((aligned(16))) unsigned short kvT[HID * 32];      // 16 KB: V^T tile
    __shared__ __attribute__((aligned(16))) unsigned short qls[4 * 16 * HID];  // 32 KB: Q bf16
    __shared__ __attribute__((aligned(16))) unsigned short pls[4 * 16 * 32];   //  4 KB: P scratch

    const int t    = threadIdx.x;
    const int lane = t & 31;
    const int w    = t >> 5;
    const int half = lane >> 4;
    const int ln16 = lane & 15;

    const int bb = blockIdx.x >> 6;          // batch
    const int qb = blockIdx.x & 63;
    const long batchbase = (long)bb * SDIM;
    const int q0 = qb * 64 + w * 16;         // query-row base within batch

    // Stage this wave's Q tile (16 x 256 bf16) into LDS
    unsigned short* q_my = &qls[w * 16 * HID];
    #pragma unroll
    for (int i = 0; i < 16; ++i) {
        uint4 v = *(const uint4*)&linb[(batchbase + q0 + i) * HID + lane * 8];
        *(uint4*)&q_my[i * HID + lane * 8] = v;
    }
    unsigned short* p_my = &pls[w * 16 * 32];

    // Constant pieces of the TDM descriptor (D#), ISA section 8.3/8.4:
    // group1: data_size=2B, tensor_dim0=256, tensor_dim1=4096, tile=256x32,
    //         tensor_dim0_stride=256, no padding / iterate / multicast.
    const unsigned lds_kv = (unsigned)(unsigned long long)(uintptr_t)&kv[0]; // LDS byte offset (addr[31:0])
    const i32x8 g1 = { 0x00010000,   // [17:16] data_size = 1 (2 bytes)
                       0x01000000,   // [79:48] tensor_dim0 = 256 (low 16 in dw1[31:16])
                       0x10000000,   // [111:80] tensor_dim1 = 4096 (low 16 in dw2[31:16])
                       0x01000000,   // [127:112] tile_dim0 = 256
                       32,           // [143:128] tile_dim1 = 32 keys
                       256,          // [207:160] tensor_dim0_stride = 256
                       0, 0 };
    const i32x4 gz  = { 0, 0, 0, 0 };            // groups 2/3 unused (2D tensor)
    const i32x8 gz8 = { 0, 0, 0, 0, 0, 0, 0, 0 }; // extra group (clang-23 6-arg form)

    float ms[8], ls[8], sc[8];
    float o[16][8];
    #pragma unroll
    for (int r = 0; r < 8; ++r) { ms[r] = -1e30f; ls[r] = 0.f; }
    #pragma unroll
    for (int nt = 0; nt < 16; ++nt)
        #pragma unroll
        for (int r = 0; r < 8; ++r) o[nt][r] = 0.f;

    for (int it = 0; it < SDIM / 32; ++it) {
        const int key0 = it * 32;
        __syncthreads();                       // previous iteration done with kv/kvT
        if (w == 0) {
            // TDM: DMA the 32x256 bf16 K/V tile into LDS (one op per tile)
            unsigned long long ga =
                (unsigned long long)(uintptr_t)&linb[(batchbase + key0) * HID];
            u32x4 g0 = { 1u,                                   // count=1 (valid D#)
                         lds_kv,                               // [63:32] lds_addr
                         (unsigned)(ga & 0xFFFFFFFFull),       // [95:64] global_addr lo
                         ((unsigned)((ga >> 32) & 0x1FFFFFFull))
                             | 0x80000000u };                  // [120:96] addr hi, [127:126] type=2
            __builtin_amdgcn_tensor_load_to_lds(g0, g1, gz, gz, gz8, 0);
            __builtin_amdgcn_s_wait_tensorcnt(0);
        }
        if (it + 1 < SDIM / 32) {              // warm L2 for the next tile
            __builtin_prefetch(&linb[(batchbase + key0 + 32 + (t >> 2)) * HID + (t & 3) * 64], 0, 1);
        }
        __syncthreads();                       // kv tile visible to all waves

        // Build kvT[col][key] (V^T) from the TDM-staged row-major tile
        for (int i = 0; i < 8; ++i) {
            int u4  = i * 128 + t;             // 1024 uint4
            int key = u4 >> 5;
            int c8  = (u4 & 31) << 3;
            uint4 v = *(const uint4*)&kv[key * HID + c8];
            const unsigned short* hp = (const unsigned short*)&v;
            #pragma unroll
            for (int h = 0; h < 8; ++h) kvT[(c8 + h) * 32 + key] = hp[h];
        }
        __syncthreads();                       // kvT ready

        // S = Q K^T for 32 keys -> two 16x16 f32 tiles (K read from LDS)
        F8 s0, s1;
        #pragma unroll
        for (int r = 0; r < 8; ++r) { s0.f[r] = 0.f; s1.f[r] = 0.f; }
        #pragma unroll
        for (int ds = 0; ds < 8; ++ds) {
            const int dbase = ds * 32;
            A16 a, b0, b1;
            a.q[0] = *(const uint4*)&q_my[ln16 * HID + dbase + half * 8];
            a.q[1] = *(const uint4*)&q_my[ln16 * HID + dbase + 16 + half * 8];
            b0.q[0] = *(const uint4*)&kv[ln16 * HID + dbase + half * 8];
            b0.q[1] = *(const uint4*)&kv[ln16 * HID + dbase + 16 + half * 8];
            b1.q[0] = *(const uint4*)&kv[(16 + ln16) * HID + dbase + half * 8];
            b1.q[1] = *(const uint4*)&kv[(16 + ln16) * HID + dbase + 16 + half * 8];
            s0.v = wmma_bf16(a.v, b0.v, s0.v);
            s1.v = wmma_bf16(a.v, b1.v, s1.v);
        }

        // Online softmax: per-row (VGPR r, lane-half) max/sum via 16-lane reductions
        #pragma unroll
        for (int r = 0; r < 8; ++r) {
            float mx = fmaxf(s0.f[r], s1.f[r]);
            #pragma unroll
            for (int m = 1; m < 16; m <<= 1) mx = fmaxf(mx, __shfl_xor(mx, m, 32));
            float nm = fmaxf(ms[r], mx);
            sc[r] = __expf(ms[r] - nm);
            float p0 = __expf(s0.f[r] - nm);
            float p1 = __expf(s1.f[r] - nm);
            s0.f[r] = p0; s1.f[r] = p1;
            float rs = p0 + p1;
            #pragma unroll
            for (int m = 1; m < 16; m <<= 1) rs += __shfl_xor(rs, m, 32);
            ls[r] = ls[r] * sc[r] + rs;
            ms[r] = nm;
        }

        // Transpose P (D-layout -> A-layout) through per-wave LDS scratch, bf16
        #pragma unroll
        for (int r = 0; r < 8; ++r) {
            int row = r + half * 8;
            p_my[row * 32 + ln16]      = f2b(s0.f[r]);
            p_my[row * 32 + 16 + ln16] = f2b(s1.f[r]);
        }
        A16 pa;
        pa.q[0] = *(const uint4*)&p_my[ln16 * 32 + half * 8];
        pa.q[1] = *(const uint4*)&p_my[ln16 * 32 + 16 + half * 8];

        // O = O*scale + P @ V  (16 column tiles)
        #pragma unroll
        for (int nt = 0; nt < 16; ++nt) {
            const int col = nt * 16 + ln16;
            A16 bv;
            bv.q[0] = *(const uint4*)&kvT[col * 32 + half * 8];
            bv.q[1] = *(const uint4*)&kvT[col * 32 + 16 + half * 8];
            F8 oc;
            #pragma unroll
            for (int r = 0; r < 8; ++r) oc.f[r] = o[nt][r] * sc[r];
            oc.v = wmma_bf16(pa.v, bv.v, oc.v);
            #pragma unroll
            for (int r = 0; r < 8; ++r) o[nt][r] = oc.f[r];
        }
    }

    // Epilogue: 1/l, residual add, LayerNorm over H=256, store
    float inv_l[8];
    #pragma unroll
    for (int r = 0; r < 8; ++r) inv_l[r] = 1.0f / ls[r];

    float psum[8], psq[8];
    #pragma unroll
    for (int r = 0; r < 8; ++r) { psum[r] = 0.f; psq[r] = 0.f; }
    #pragma unroll
    for (int nt = 0; nt < 16; ++nt) {
        const int col = nt * 16 + ln16;
        #pragma unroll
        for (int r = 0; r < 8; ++r) {
            long grow = batchbase + q0 + r + half * 8;
            float v = o[nt][r] * inv_l[r] + linf[grow * HID + col];
            o[nt][r] = v;
            psum[r] += v;
            psq[r]  += v * v;
        }
    }
    #pragma unroll
    for (int r = 0; r < 8; ++r) {
        #pragma unroll
        for (int m = 1; m < 16; m <<= 1) {
            psum[r] += __shfl_xor(psum[r], m, 32);
            psq[r]  += __shfl_xor(psq[r],  m, 32);
        }
    }
    #pragma unroll
    for (int nt = 0; nt < 16; ++nt) {
        const int col = nt * 16 + ln16;
        const float g  = gamma[col];
        const float bt = beta[col];
        #pragma unroll
        for (int r = 0; r < 8; ++r) {
            float mu   = psum[r] * (1.0f / 256.0f);
            float var  = fmaxf(psq[r] * (1.0f / 256.0f) - mu * mu, 0.0f);
            float rstd = rsqrtf(var + 1e-5f);
            long grow  = batchbase + q0 + r + half * 8;
            out[grow * HID + col] = (o[nt][r] - mu) * rstd * g + bt;
        }
    }
}

// ---------------------------------------------------------------------------
extern "C" void kernel_launch(void* const* d_in, const int* in_sizes, int n_in,
                              void* d_out, int out_size, void* d_ws, size_t ws_size,
                              hipStream_t stream) {
    const float* x     = (const float*)d_in[0];   // [8,4096,512]
    const float* W     = (const float*)d_in[1];   // [512,256]
    const float* bias  = (const float*)d_in[2];   // [256]
    const float* gamma = (const float*)d_in[3];   // [256]
    const float* beta  = (const float*)d_in[4];   // [256]
    float* out = (float*)d_out;                   // [8,4096,256] f32

    char* ws = (char*)d_ws;
    size_t off = 0;
    float* linf = (float*)(ws + off);            off += (size_t)BATCH * SDIM * HID * sizeof(float);          // 32 MB
    unsigned short* linb = (unsigned short*)(ws + off); off += (size_t)BATCH * SDIM * HID * sizeof(unsigned short); // 16 MB
    unsigned short* wt   = (unsigned short*)(ws + off); off += (size_t)DIN * HID * sizeof(unsigned short);    // 256 KB

    k_wt    <<<(DIN * HID + 255) / 256, 256, 0, stream>>>(W, wt);
    k_linear<<<BATCH * SDIM / 16,        128, 0, stream>>>(x, wt, bias, linf, linb);
    k_attn  <<<BATCH * (SDIM / 64),      128, 0, stream>>>(linf, linb, gamma, beta, out);
}
